// LocalSelfAttention_77635828842928
// MI455X (gfx1250) — matrix operand order repaired and checked by reading, not compiled, not run on hardware
//
#include <hip/hip_runtime.h>

typedef __attribute__((ext_vector_type(16))) __bf16          v16bf;
typedef __attribute__((ext_vector_type(16))) unsigned short  v16us;
typedef __attribute__((ext_vector_type(8)))  unsigned short  v8us;
typedef __attribute__((ext_vector_type(8)))  float           v8f;

#define T_SEQ   2048
#define H_DIM   2048
#define N_HEADS 16
#define D_HEAD  128
#define K_BLK   50
#define N_BLK   41
#define NEG_VAL (-1.0e9f)
#define Q_SCALE 0.08838834764831845f   // 1/sqrt(128)

// LDS row strides (u16 units); all multiples of 8 so per-lane 16B chunks are
// 16-byte aligned -> ds_load_b128.
#define LDA_G  40   // GEMM A / Bt tiles (32 cols)
#define LDA_QK 136  // attention Q/K tiles (128 cols)
#define LDA_PV 72   // attention P / Vt tiles (64 cols)

// ---------- bf16 helpers (round-to-nearest-even) ----------
__device__ __forceinline__ unsigned short f2bf_bits(float f) {
  unsigned u = __builtin_bit_cast(unsigned, f);
  unsigned r = u + 0x7FFFu + ((u >> 16) & 1u);
  return (unsigned short)(r >> 16);
}

// ---------- WMMA fragment gathers from LDS (layouts per ISA 7.12.2) ----------
// A (16x32 bf16, row-major src, 16B-aligned rows): lane m=l&15 holds
// K = {hi..hi+7} U {16+hi..16+hi+7}, hi = 8*(lane>=16). Two b128 loads.
__device__ __forceinline__ v16bf load_a_frag(const unsigned short* src, int ld, int lane) {
  const unsigned short* row = src + (lane & 15) * ld + ((lane & 16) ? 8 : 0);
  v8us lo = *(const v8us*)(row);        // k = hi+0..7   (elements 0..7)
  v8us hi = *(const v8us*)(row + 16);   // k = 16+hi+0..7 (elements 8..15)
  v16us c = __builtin_shufflevector(lo, hi, 0, 1, 2, 3, 4, 5, 6, 7,
                                    8, 9, 10, 11, 12, 13, 14, 15);
  return __builtin_bit_cast(v16bf, c);
}
// B (32x16 bf16) gathered from a K-transposed image srcT[n*ld + k]:
// lane holds N=lane&15, K = {hi..hi+7} U {16+hi..16+hi+7}, hi = 8*(lane>=16).
__device__ __forceinline__ v16bf load_b_frag(const unsigned short* srcT, int ld, int lane) {
  const unsigned short* col = srcT + (lane & 15) * ld + ((lane & 16) ? 8 : 0);
  v8us lo = *(const v8us*)(col);
  v8us hi = *(const v8us*)(col + 16);
  v16us c = __builtin_shufflevector(lo, hi, 0, 1, 2, 3, 4, 5, 6, 7,
                                    8, 9, 10, 11, 12, 13, 14, 15);
  return __builtin_bit_cast(v16bf, c);
}

#define WMMA_BF16(A, B, C) \
  __builtin_amdgcn_wmma_f32_16x16x32_bf16(false, (A), false, (B), (short)0, (C), false, false)

// ---------- Generic 128x128-tile bf16 WMMA GEMM ----------
// C[MxN] = A[MxK] * B[KxN] * scale.  A: f32 or bf16 (converted on stage-in),
// B: f32 (converted on stage-in), C: f32 or bf16.  M,N,K multiples of 128/32.
template <bool A_F32, bool OUT_F32>
__global__ __launch_bounds__(256) void gemm_wmma(const void* __restrict__ Ap,
                                                 const float* __restrict__ Bp,
                                                 void* __restrict__ Cp,
                                                 int M, int N, int Kd, float scale) {
  __shared__ __attribute__((aligned(16))) unsigned short As[128 * LDA_G];
  __shared__ __attribute__((aligned(16))) unsigned short Bt[128 * LDA_G];
  const int tid = threadIdx.x;
  const int lane = tid & 31;
  const int w = tid >> 5;
  const int wr = w & 3;    // 4 row strips of 32
  const int wc = w >> 2;   // 2 col strips of 64
  const int m0 = blockIdx.y * 128, n0 = blockIdx.x * 128;

  v8f acc[2][4];
#pragma unroll
  for (int i = 0; i < 2; ++i)
#pragma unroll
    for (int j = 0; j < 4; ++j) acc[i][j] = (v8f){};

  for (int k0 = 0; k0 < Kd; k0 += 32) {
    // stage A tile 128x32 (row-major)
#pragma unroll
    for (int i = tid; i < 128 * 32; i += 256) {
      int r = i >> 5, c = i & 31;
      size_t g = (size_t)(m0 + r) * Kd + (k0 + c);
      As[r * LDA_G + c] = A_F32 ? f2bf_bits(((const float*)Ap)[g])
                                : ((const unsigned short*)Ap)[g];
    }
    // stage B tile 32x128 transposed -> Bt[n][k]
#pragma unroll
    for (int i = tid; i < 32 * 128; i += 256) {
      int k = i >> 7, nn = i & 127;
      Bt[nn * LDA_G + k] = f2bf_bits(Bp[(size_t)(k0 + k) * N + (n0 + nn)]);
    }
    // prefetch next k-tile into the cache hierarchy (global_prefetch_b8)
    if (k0 + 32 < Kd) {
      if (A_F32)
        __builtin_prefetch((const float*)Ap + (size_t)(m0 + (tid >> 1)) * Kd + (k0 + 32), 0, 3);
      else
        __builtin_prefetch((const unsigned short*)Ap + (size_t)(m0 + (tid >> 1)) * Kd + (k0 + 32), 0, 3);
      __builtin_prefetch(Bp + (size_t)(k0 + 32 + (tid >> 4)) * N + n0 + ((tid & 15) << 3), 0, 3);
    }
    __syncthreads();
    v16bf a0 = load_a_frag(As + (wr * 32) * LDA_G, LDA_G, lane);
    v16bf a1 = load_a_frag(As + (wr * 32 + 16) * LDA_G, LDA_G, lane);
#pragma unroll
    for (int ct = 0; ct < 4; ++ct) {
      v16bf b = load_b_frag(Bt + (wc * 64 + ct * 16) * LDA_G, LDA_G, lane);
      acc[0][ct] = WMMA_BF16(a0, b, acc[0][ct]);
      acc[1][ct] = WMMA_BF16(a1, b, acc[1][ct]);
    }
    __syncthreads();
  }
  // epilogue: C/D layout -> lanes 0-15 rows r, lanes 16-31 rows r+8; N=lane&15
  const int nloc = lane & 15;
  const int hi8 = (lane & 16) ? 8 : 0;
#pragma unroll
  for (int mt = 0; mt < 2; ++mt)
#pragma unroll
    for (int ct = 0; ct < 4; ++ct)
#pragma unroll
      for (int r = 0; r < 8; ++r) {
        int row = m0 + wr * 32 + mt * 16 + r + hi8;
        int col = n0 + wc * 64 + ct * 16 + nloc;
        float vv = acc[mt][ct][r] * scale;
        if (OUT_F32) ((float*)Cp)[(size_t)row * N + col] = vv;
        else ((unsigned short*)Cp)[(size_t)row * N + col] = f2bf_bits(vv);
      }
}

// ---------- Block-local attention: one workgroup per (block n, head) ----------
// 128 threads = 4 waves; wave w owns query rows [w*16, w*16+16) of the 64-row
// padded block (K_BLK=50 valid rows).
__global__ __launch_bounds__(128) void attn_kernel(const unsigned short* __restrict__ Qb,
                                                   const unsigned short* __restrict__ Kb,
                                                   const unsigned short* __restrict__ Vb,
                                                   const float* __restrict__ bias,
                                                   const float* __restrict__ mask,
                                                   unsigned short* __restrict__ Ob) {
  __shared__ __attribute__((aligned(16))) unsigned short Qs[64 * LDA_QK];
  __shared__ __attribute__((aligned(16))) unsigned short Ks[64 * LDA_QK];
  __shared__ __attribute__((aligned(16))) unsigned short Vt[128 * LDA_PV];
  __shared__ __attribute__((aligned(16))) unsigned short Ps[64 * LDA_PV];
  const int nb = blockIdx.x;
  const int head = blockIdx.y;
  const int tid = threadIdx.x;
  const int lane = tid & 31;
  const int w = tid >> 5;
  const int t0 = nb * K_BLK;

  // Stage Q,K row-major and V transposed; zero-fill pad rows.
  for (int i = tid; i < 64 * 128; i += 128) {
    int r = i >> 7, d = i & 127;
    int t = t0 + r;
    bool valid = (r < K_BLK) && (t < T_SEQ);
    size_t g = (size_t)t * H_DIM + (size_t)head * D_HEAD + d;
    unsigned short q = valid ? Qb[g] : (unsigned short)0;
    unsigned short k = valid ? Kb[g] : (unsigned short)0;
    unsigned short v = valid ? Vb[g] : (unsigned short)0;
    Qs[r * LDA_QK + d] = q;
    Ks[r * LDA_QK + d] = k;
    Vt[d * LDA_PV + r] = v;
  }
  __syncthreads();

  // S strip (16x64) = Q strip (16x128) @ K^T; Ks row-major IS K^T image for B-frag.
  v16bf aQ[4];
#pragma unroll
  for (int kk = 0; kk < 4; ++kk)
    aQ[kk] = load_a_frag(Qs + (w * 16) * LDA_QK + kk * 32, LDA_QK, lane);
  v8f S[4];
#pragma unroll
  for (int ct = 0; ct < 4; ++ct) {
    v8f acc = (v8f){};
#pragma unroll
    for (int kk = 0; kk < 4; ++kk) {
      v16bf b = load_b_frag(Ks + (ct * 16) * LDA_QK + kk * 32, LDA_QK, lane);
      acc = WMMA_BF16(aQ[kk], b, acc);
    }
    S[ct] = acc;
  }

  const int nloc = lane & 15;
  const int hi8 = (lane & 16) ? 8 : 0;
  float mq[8];
  int tq[8];
#pragma unroll
  for (int r = 0; r < 8; ++r) {
    int m = w * 16 + r + hi8;
    int t = t0 + m;
    tq[r] = t;
    mq[r] = (m < K_BLK && t < T_SEQ) ? mask[t] : NEG_VAL;
  }
  // logits += diag-block bias + pairwise mask; hard-exclude pad cols >= 50
#pragma unroll
  for (int ct = 0; ct < 4; ++ct) {
    int c = ct * 16 + nloc;
    int tk = t0 + c;
    float mk = (c < K_BLK && tk < T_SEQ) ? mask[tk] : NEG_VAL;
#pragma unroll
    for (int r = 0; r < 8; ++r) {
      float l;
      if (c >= K_BLK) {
        l = -1.0e30f;
      } else {
        int m = w * 16 + r + hi8;
        float bv = (m < K_BLK && tq[r] < T_SEQ && tk < T_SEQ)
                       ? bias[(size_t)head * T_SEQ * T_SEQ + (size_t)tq[r] * T_SEQ + tk]
                       : 0.0f;
        l = S[ct][r] + bv + ((mq[r] * mk > 0.0f) ? 0.0f : NEG_VAL);
      }
      S[ct][r] = l;
    }
  }
  // Row softmax: row r lives in one 16-lane half across 4 tiles.
#pragma unroll
  for (int r = 0; r < 8; ++r) {
    float mx = fmaxf(fmaxf(S[0][r], S[1][r]), fmaxf(S[2][r], S[3][r]));
#pragma unroll
    for (int off = 1; off <= 8; off <<= 1) mx = fmaxf(mx, __shfl_xor(mx, off, 32));
    float sum = 0.0f;
#pragma unroll
    for (int ct = 0; ct < 4; ++ct) {
      float e = __expf(S[ct][r] - mx);
      S[ct][r] = e;
      sum += e;
    }
#pragma unroll
    for (int off = 1; off <= 8; off <<= 1) sum += __shfl_xor(sum, off, 32);
    float inv = 1.0f / sum;
#pragma unroll
    for (int ct = 0; ct < 4; ++ct) S[ct][r] *= inv;
  }
  // Redistribute P (C-layout -> A-layout) via per-wave LDS strip.
#pragma unroll
  for (int ct = 0; ct < 4; ++ct)
#pragma unroll
    for (int r = 0; r < 8; ++r)
      Ps[(w * 16 + r + hi8) * LDA_PV + ct * 16 + nloc] = f2bf_bits(S[ct][r]);

  // O strip (16x128) = P (16x64) @ V (64x128); Vt[d][c] is the B-frag image.
  v16bf aP0 = load_a_frag(Ps + (w * 16) * LDA_PV, LDA_PV, lane);
  v16bf aP1 = load_a_frag(Ps + (w * 16) * LDA_PV + 32, LDA_PV, lane);
#pragma unroll
  for (int dt = 0; dt < 8; ++dt) {
    v8f acc = (v8f){};
    v16bf b0 = load_b_frag(Vt + (dt * 16) * LDA_PV, LDA_PV, lane);
    acc = WMMA_BF16(aP0, b0, acc);
    v16bf b1 = load_b_frag(Vt + (dt * 16) * LDA_PV + 32, LDA_PV, lane);
    acc = WMMA_BF16(aP1, b1, acc);
#pragma unroll
    for (int r = 0; r < 8; ++r) {
      int m = w * 16 + r + hi8;
      int t = t0 + m;
      if (m < K_BLK && t < T_SEQ)
        Ob[(size_t)t * H_DIM + (size_t)head * D_HEAD + dt * 16 + nloc] = f2bf_bits(acc[r]);
    }
  }
}

extern "C" void kernel_launch(void* const* d_in, const int* in_sizes, int n_in,
                              void* d_out, int out_size, void* d_ws, size_t ws_size,
                              hipStream_t stream) {
  (void)in_sizes; (void)n_in; (void)out_size; (void)ws_size;
  const float* x    = (const float*)d_in[0];  // (1,2048,2048)
  const float* mask = (const float*)d_in[1];  // (1,2048)
  const float* bias = (const float*)d_in[2];  // (1,16,2048,2048)
  const float* Wq   = (const float*)d_in[3];  // (2048,16,128) -> 2048x2048
  const float* Wk   = (const float*)d_in[4];
  const float* Wv   = (const float*)d_in[5];
  const float* Wo   = (const float*)d_in[6];  // (16,128,2048) -> 2048x2048

  const size_t MAT = (size_t)T_SEQ * H_DIM;   // 4M elems
  unsigned short* Qb = (unsigned short*)d_ws;
  unsigned short* Kb = Qb + MAT;
  unsigned short* Vb = Kb + MAT;
  unsigned short* Ob = Vb + MAT;              // total 32 MB of bf16 scratch

  dim3 gg(H_DIM / 128, T_SEQ / 128);
  gemm_wmma<true, false><<<gg, 256, 0, stream>>>(x, Wq, Qb, T_SEQ, H_DIM, H_DIM, Q_SCALE);
  gemm_wmma<true, false><<<gg, 256, 0, stream>>>(x, Wk, Kb, T_SEQ, H_DIM, H_DIM, 1.0f);
  gemm_wmma<true, false><<<gg, 256, 0, stream>>>(x, Wv, Vb, T_SEQ, H_DIM, H_DIM, 1.0f);
  attn_kernel<<<dim3(N_BLK, N_HEADS), 128, 0, stream>>>(Qb, Kb, Vb, bias, mask, Ob);
  gemm_wmma<false, true><<<gg, 256, 0, stream>>>(Ob, Wo, d_out, T_SEQ, H_DIM, H_DIM, 1.0f);
}
